// MySuperNetLittleInput_76459007803615
// MI455X (gfx1250) — matrix operand order repaired and checked by reading, not compiled, not run on hardware
//
#include <hip/hip_runtime.h>
#include <hip/hip_bf16.h>
#include <math.h>

#define HH 96
#define NN (HH * HH) /* 9216 spatial positions */

typedef __attribute__((ext_vector_type(16))) __bf16 v16bf;
typedef __attribute__((ext_vector_type(8)))  __bf16 v8bf;
typedef __attribute__((ext_vector_type(8)))  float  v8f;

__device__ __forceinline__ v8f wmma_bf16(v16bf a, v16bf b, v8f c) {
  // D(16x16,f32) = A(16x32,bf16) x B(32x16,bf16) + C
  return __builtin_amdgcn_wmma_f32_16x16x32_bf16(false, a, false, b, (short)0, c,
                                                 false, false);
}

// ---------------------------------------------------------------------------
// im2col into bf16, fragment-ready layout: col[n][Kpad], k = ci*9 + tap
// (matches OIHW weight flattening). Zero-padded to Kpad (multiple of 32).
// All the divides / boundary branches happen HERE, once, in a bulk pass,
// instead of inside the WMMA inner loop.
// ---------------------------------------------------------------------------
__global__ void im2col_kernel(const float* __restrict__ in,
                              __bf16* __restrict__ col, int Cin, int Kpad) {
  int k = blockIdx.x * blockDim.x + threadIdx.x;
  int n = blockIdx.y;
  if (k >= Kpad) return;
  int Ktot = Cin * 9;
  float v = 0.f;
  if (k < Ktot) {
    int ci = k / 9, tap = k - ci * 9;
    int y0 = n / HH, x0 = n - y0 * HH;
    int yy = y0 + tap / 3 - 1;
    int xx = x0 + tap % 3 - 1;
    if (yy >= 0 && yy < HH && xx >= 0 && xx < HH)
      v = in[(size_t)ci * NN + yy * HH + xx];
  }
  col[(size_t)n * Kpad + k] = (__bf16)v;
}

// Weights -> bf16, padded [CoutPad][Kpad] (zeros beyond Cout/Ktot).
__global__ void wcvt_kernel(const float* __restrict__ w, __bf16* __restrict__ wb,
                            int Ktot, int Kpad, int Cout) {
  int k = blockIdx.x * blockDim.x + threadIdx.x;
  int co = blockIdx.y;
  if (k >= Kpad) return;
  float v = (k < Ktot && co < Cout) ? w[(size_t)co * Ktot + k] : 0.f;
  wb[(size_t)co * Kpad + k] = (__bf16)v;
}

// ---------------------------------------------------------------------------
// Clean WMMA GEMM: out[co][n] = sum_k col[n][k] * wb[co][k].
// 256 threads = 8 waves; each wave owns one 16(spatial) x 32(cout) tile:
// one A fragment (two 16B contiguous loads) reused across two B fragments
// (one 32B contiguous load each) -> 2 WMMAs per K-chunk, zero branches.
// Also accumulates LayerNorm sum/sumsq via wave-reduced atomics.
// ---------------------------------------------------------------------------
__global__ void gemm_wmma_kernel(const __bf16* __restrict__ col,
                                 const __bf16* __restrict__ wb,
                                 float* __restrict__ out,
                                 float* __restrict__ stats,
                                 int Kpad, int Cout) {
  const int wave = threadIdx.x >> 5;
  const int lane = threadIdx.x & 31;
  const int half = lane >> 4;
  const int lm   = lane & 15;
  const int mtile = blockIdx.x * 8 + wave;          // 0..575
  const int co0 = blockIdx.y * 32 + lm;             // B tile 0 column
  const int co1 = co0 + 16;                         // B tile 1 column

  const __bf16* arow  = col + (size_t)(mtile * 16 + lm) * Kpad;
  const __bf16* b0row = wb + (size_t)co0 * Kpad + 16 * half;
  const __bf16* b1row = wb + (size_t)co1 * Kpad + 16 * half;

  v8f acc0 = {}, acc1 = {};
  for (int kb = 0; kb < Kpad; kb += 32) {
    // A layout: elems -> K = kb + 16*half + i (contiguous)
    v8bf lo = *(const v8bf*)(arow + kb + 16 * half);
    v8bf hi = *(const v8bf*)(arow + kb + 16 * half + 8);
    v16bf a;
#pragma unroll
    for (int i = 0; i < 8; ++i) { a[i] = lo[i]; a[i + 8] = hi[i]; }
    // B layout: elems -> K = kb + 16*half + i (contiguous 32B)
    v16bf b0 = *(const v16bf*)(b0row + kb);
    v16bf b1 = *(const v16bf*)(b1row + kb);
    acc0 = wmma_bf16(a, b0, acc0);
    acc1 = wmma_bf16(a, b1, acc1);
  }

  float s = 0.f, s2 = 0.f;
#pragma unroll
  for (int r = 0; r < 8; ++r) {   // D row = r + 8*half (spatial)
    int n = mtile * 16 + r + 8 * half;
    if (co0 < Cout) { float v = acc0[r]; out[(size_t)co0 * NN + n] = v; s += v; s2 += v * v; }
    if (co1 < Cout) { float v = acc1[r]; out[(size_t)co1 * NN + n] = v; s += v; s2 += v * v; }
  }
#pragma unroll
  for (int off = 16; off > 0; off >>= 1) {
    s  += __shfl_xor(s,  off, 32);
    s2 += __shfl_xor(s2, off, 32);
  }
  if (lane == 0) { atomicAdd(&stats[0], s); atomicAdd(&stats[1], s2); }
}

// ---------------------------------------------------------------------------
// LayerNorm over whole tensor (B=1) + per-element affine + ReLU, in place.
// ---------------------------------------------------------------------------
__global__ void ln_relu_kernel(float* __restrict__ data,
                               const float* __restrict__ lw,
                               const float* __restrict__ lb,
                               const float* __restrict__ stats, int total) {
  int i = blockIdx.x * blockDim.x + threadIdx.x;
  if (i >= total) return;
  float inv = 1.f / (float)total;
  float mean = stats[0] * inv;
  float var  = stats[1] * inv - mean * mean;
  float rs = rsqrtf(var + 1e-5f);
  float v = (data[i] - mean) * rs * lw[i] + lb[i];
  data[i] = v > 0.f ? v : 0.f;
}

// ---------------------------------------------------------------------------
// CAM gram: gram[c][d] = sum_n h[c,n] h[d,n]  (17x17, negligible FLOPs)
// ---------------------------------------------------------------------------
__global__ void gram_kernel(const float* __restrict__ h, float* __restrict__ gram) {
  int cd = blockIdx.x;  // 0..288
  int c = cd / 17, d = cd % 17;
  float s = 0.f;
  for (int n = threadIdx.x; n < NN; n += blockDim.x)
    s += h[(size_t)c * NN + n] * h[(size_t)d * NN + n];
  __shared__ float red[8];
#pragma unroll
  for (int off = 16; off > 0; off >>= 1) s += __shfl_xor(s, off, 32);
  int wid = threadIdx.x >> 5;
  if ((threadIdx.x & 31) == 0) red[wid] = s;
  __syncthreads();
  if (threadIdx.x == 0) {
    float t = 0.f;
    int nw = blockDim.x >> 5;
    for (int wv = 0; wv < nw; ++wv) t += red[wv];
    gram[cd] = t;
  }
}

// attn = softmax(rowmax(e) - e) row-wise  (== softmax(rowmin - e))
__global__ void cam_softmax_kernel(const float* __restrict__ gram,
                                   float* __restrict__ attn) {
  int c = threadIdx.x;
  if (c >= 17) return;
  float e[17], mn = INFINITY;
  for (int d = 0; d < 17; ++d) { e[d] = gram[c * 17 + d]; mn = fminf(mn, e[d]); }
  float sum = 0.f, p[17];
  for (int d = 0; d < 17; ++d) { p[d] = __expf(mn - e[d]); sum += p[d]; }
  float is = 1.f / sum;
  for (int d = 0; d < 17; ++d) attn[c * 17 + d] = p[d] * is;
}

// ---------------------------------------------------------------------------
// q/k/v 1x1 convs with bias, emitted in WMMA-fragment-ready bf16 layouts:
//   qT,kT: [N][32] channel-padded rows ; vB: [32][N] row per channel.
// ---------------------------------------------------------------------------
__global__ void qkv_kernel(const float* __restrict__ h,
                           const float* __restrict__ qw, const float* __restrict__ qb,
                           const float* __restrict__ kw, const float* __restrict__ kb,
                           const float* __restrict__ vw, const float* __restrict__ vb,
                           __bf16* __restrict__ qT, __bf16* __restrict__ kT,
                           __bf16* __restrict__ vB) {
  int n = blockIdx.x * blockDim.x + threadIdx.x;
  if (n >= NN) return;
  float hv[17];
#pragma unroll
  for (int d = 0; d < 17; ++d) hv[d] = h[(size_t)d * NN + n];
  for (int c = 0; c < 32; ++c) {
    float q = 0.f, k = 0.f, v = 0.f;
    if (c < 17) {
      q = qb[c]; k = kb[c]; v = vb[c];
#pragma unroll
      for (int d = 0; d < 17; ++d) {
        q += qw[c * 17 + d] * hv[d];
        k += kw[c * 17 + d] * hv[d];
        v += vw[c * 17 + d] * hv[d];
      }
    }
    qT[(size_t)n * 32 + c] = (__bf16)q;
    kT[(size_t)n * 32 + c] = (__bf16)k;
    vB[(size_t)c * NN + n] = (__bf16)v;
  }
}

// ---------------------------------------------------------------------------
// Flash-style PAM. 8 waves/block, one 16-query tile per wave. Per key chunk
// of 32: energy (2 WMMA, K = 32 padded channels) -> online softmax (half-wave
// shfl reductions; D rows 0-7 in lanes 0-15, rows 8-15 in lanes 16-31) ->
// restage P via per-wave LDS into A layout -> PV (2 WMMA, K = 32 keys).
// The 9216x9216 energy matrix (340MB f32, > L2) never touches memory.
// ---------------------------------------------------------------------------
__global__ void flash_pam_kernel(const __bf16* __restrict__ qT,
                                 const __bf16* __restrict__ kT,
                                 const __bf16* __restrict__ vB,
                                 float* __restrict__ pam) {
  const int wave = threadIdx.x >> 5;
  const int lane = threadIdx.x & 31;
  const int half = lane >> 4, lm = lane & 15;
  const int qbase = (blockIdx.x * 8 + wave) * 16;
  __shared__ float Pbuf[8][16 * 32];
  float* P = Pbuf[wave];

  v16bf aq;  // Q fragment: A[mq][c], loaded once
  {
    const __bf16* qrow = qT + (size_t)(qbase + lm) * 32;
#pragma unroll
    for (int i = 0; i < 16; ++i) {
      int c = i + 16 * half;
      aq[i] = qrow[c];
    }
  }
  float mrow[8], lrow[8];
#pragma unroll
  for (int r = 0; r < 8; ++r) { mrow[r] = -INFINITY; lrow[r] = 0.f; }
  v8f o0 = {}, o1 = {};

  for (int mb = 0; mb < NN / 32; ++mb) {
    const int mbase = mb * 32;
    // K fragments: B[c][key], contiguous 32B vector load per lane
    v16bf bk0 = *(const v16bf*)(kT + (size_t)(mbase + lm) * 32 + 16 * half);
    v16bf bk1 = *(const v16bf*)(kT + (size_t)(mbase + 16 + lm) * 32 + 16 * half);
    v8f z = {};
    v8f e0 = wmma_bf16(aq, bk0, z);  // queries x keys[0..15]
    v8f e1 = wmma_bf16(aq, bk1, z);  // queries x keys[16..31]

#pragma unroll
    for (int r = 0; r < 8; ++r) {    // D row = r + 8*half (query)
      float mx = fmaxf(e0[r], e1[r]);
#pragma unroll
      for (int off = 8; off > 0; off >>= 1) mx = fmaxf(mx, __shfl_xor(mx, off, 32));
      float mnew  = fmaxf(mrow[r], mx);
      float scale = __expf(mrow[r] - mnew);
      float p0 = __expf(e0[r] - mnew);
      float p1 = __expf(e1[r] - mnew);
      float ps = p0 + p1;
#pragma unroll
      for (int off = 8; off > 0; off >>= 1) ps += __shfl_xor(ps, off, 32);
      lrow[r] = lrow[r] * scale + ps;
      mrow[r] = mnew;
      o0[r] *= scale; o1[r] *= scale;
      P[(r + 8 * half) * 32 + lm]      = p0;
      P[(r + 8 * half) * 32 + lm + 16] = p1;
    }
    __syncthreads();
    v16bf ap;  // P fragment: A[query][key_local]
#pragma unroll
    for (int i = 0; i < 16; ++i) {
      int k = i + 16 * half;
      ap[i] = (__bf16)P[lm * 32 + k];
    }
    // V fragments: B[key_local][c], contiguous 32B vector load per lane
    v16bf bv0 = *(const v16bf*)(vB + (size_t)lm * NN + mbase + 16 * half);
    v16bf bv1 = *(const v16bf*)(vB + (size_t)(lm + 16) * NN + mbase + 16 * half);
    o0 = wmma_bf16(ap, bv0, o0);
    o1 = wmma_bf16(ap, bv1, o1);
    __syncthreads();
  }
#pragma unroll
  for (int r = 0; r < 8; ++r) {
    int n = qbase + r + 8 * half;
    float inv = 1.f / lrow[r];
    pam[(size_t)lm * NN + n] = o0[r] * inv;                 // c = 0..15
    if (lm == 0) pam[(size_t)16 * NN + n] = o1[r] * inv;    // c = 16 (rest pad)
  }
}

// ---------------------------------------------------------------------------
// y = gamma_c*(attn @ h) + gamma_p*pam + h, fused with conv5 1x1 + LN stats.
// ---------------------------------------------------------------------------
__global__ void combine_conv5_kernel(const float* __restrict__ h,
                                     const float* __restrict__ attn,
                                     const float* __restrict__ pam,
                                     const float* __restrict__ camg,
                                     const float* __restrict__ pamg,
                                     const float* __restrict__ w5,
                                     float* __restrict__ c5,
                                     float* __restrict__ stats) {
  int n = blockIdx.x * blockDim.x + threadIdx.x;
  float s = 0.f, s2 = 0.f;
  if (n < NN) {
    float g_cam = camg[0], g_pam = pamg[0];
    float hv[17], yv[17];
#pragma unroll
    for (int d = 0; d < 17; ++d) hv[d] = h[(size_t)d * NN + n];
#pragma unroll
    for (int c = 0; c < 17; ++c) {
      float cam = 0.f;
#pragma unroll
      for (int d = 0; d < 17; ++d) cam += attn[c * 17 + d] * hv[d];
      yv[c] = g_cam * cam + g_pam * pam[(size_t)c * NN + n] + hv[c];
    }
#pragma unroll
    for (int c = 0; c < 17; ++c) {
      float o = 0.f;
#pragma unroll
      for (int d = 0; d < 17; ++d) o += w5[c * 17 + d] * yv[d];
      c5[(size_t)c * NN + n] = o;
      s += o; s2 += o * o;
    }
  }
#pragma unroll
  for (int off = 16; off > 0; off >>= 1) {
    s  += __shfl_xor(s,  off, 32);
    s2 += __shfl_xor(s2, off, 32);
  }
  __shared__ float rs[8], rs2[8];
  int wid = threadIdx.x >> 5;
  if ((threadIdx.x & 31) == 0) { rs[wid] = s; rs2[wid] = s2; }
  __syncthreads();
  if (threadIdx.x == 0) {
    float a = 0.f, b = 0.f;
    int nw = blockDim.x >> 5;
    for (int wv = 0; wv < nw; ++wv) { a += rs[wv]; b += rs2[wv]; }
    atomicAdd(&stats[0], a);
    atomicAdd(&stats[1], b);
  }
}

__global__ void conv6_kernel(const float* __restrict__ y5,
                             const float* __restrict__ w6,
                             float* __restrict__ out) {
  int n = blockIdx.x * blockDim.x + threadIdx.x;
  if (n >= NN) return;
  float yv[17];
#pragma unroll
  for (int d = 0; d < 17; ++d) yv[d] = y5[(size_t)d * NN + n];
#pragma unroll
  for (int c = 0; c < 17; ++c) {
    float o = 0.f;
#pragma unroll
    for (int d = 0; d < 17; ++d) o += w6[c * 17 + d] * yv[d];
    out[(size_t)c * NN + n] = o;
  }
}

// ---------------------------------------------------------------------------
extern "C" void kernel_launch(void* const* d_in, const int* in_sizes, int n_in,
                              void* d_out, int out_size, void* d_ws, size_t ws_size,
                              hipStream_t stream) {
  (void)in_sizes; (void)n_in; (void)out_size; (void)ws_size;
  const float* x    = (const float*)d_in[0];
  const float* w1   = (const float*)d_in[1];
  const float* l1w  = (const float*)d_in[2];
  const float* l1b  = (const float*)d_in[3];
  const float* w2   = (const float*)d_in[4];
  const float* l2w  = (const float*)d_in[5];
  const float* l2b  = (const float*)d_in[6];
  const float* w3   = (const float*)d_in[7];
  const float* l3w  = (const float*)d_in[8];
  const float* l3b  = (const float*)d_in[9];
  const float* w4   = (const float*)d_in[10];
  const float* l4w  = (const float*)d_in[11];
  const float* l4b  = (const float*)d_in[12];
  const float* camg = (const float*)d_in[13];
  const float* qw   = (const float*)d_in[14];
  const float* qb   = (const float*)d_in[15];
  const float* kw   = (const float*)d_in[16];
  const float* kb   = (const float*)d_in[17];
  const float* vw   = (const float*)d_in[18];
  const float* vb   = (const float*)d_in[19];
  const float* pamg = (const float*)d_in[20];
  const float* w5   = (const float*)d_in[21];
  const float* l5w  = (const float*)d_in[22];
  const float* l5b  = (const float*)d_in[23];
  const float* w6   = (const float*)d_in[24];
  float* out = (float*)d_out;

  char* ws = (char*)d_ws;
  size_t off = 0;
  auto carve = [&](size_t bytes) -> char* {
    char* p = ws + off;
    off = (off + bytes + 255) & ~(size_t)255;
    return p;
  };
  float*  H1    = (float*)carve((size_t)96 * NN * 4);
  float*  H2    = (float*)carve((size_t)48 * NN * 4);
  float*  H3    = (float*)carve((size_t)48 * NN * 4);
  float*  H4    = (float*)carve((size_t)17 * NN * 4);
  float*  PAM   = (float*)carve((size_t)17 * NN * 4);
  float*  C5    = (float*)carve((size_t)17 * NN * 4);
  __bf16* QT    = (__bf16*)carve((size_t)NN * 32 * 2);
  __bf16* KT    = (__bf16*)carve((size_t)NN * 32 * 2);
  __bf16* VB    = (__bf16*)carve((size_t)32 * NN * 2);
  __bf16* COL   = (__bf16*)carve((size_t)NN * 2144 * 2);  // max Kpad (conv1)
  __bf16* WB    = (__bf16*)carve((size_t)96 * 2144 * 2);  // max CoutPad*Kpad
  float*  GRAM  = (float*)carve(289 * 4);
  float*  ATTN  = (float*)carve(289 * 4);
  float*  STATS = (float*)carve(16 * 4);

  hipMemsetAsync(STATS, 0, 16 * sizeof(float), stream);

  // One conv layer = im2col + weight-cvt + WMMA GEMM (+LN stats) + LN/ReLU.
  struct Layer { const float* in; const float* w; float* out;
                 const float* lw; const float* lb;
                 int Cin, Cout, CoutPad, Kpad; float* st; };
  const Layer layers[4] = {
    { x,  w1, H1, l1w, l1b, 237, 96, 96, 2144, STATS + 0 },
    { H1, w2, H2, l2w, l2b,  96, 48, 64,  864, STATS + 2 },
    { H2, w3, H3, l3w, l3b,  48, 48, 64,  448, STATS + 4 },
    { H3, w4, H4, l4w, l4b,  48, 17, 32,  448, STATS + 6 },
  };
  for (int li = 0; li < 4; ++li) {
    const Layer& L = layers[li];
    int Ktot = L.Cin * 9;
    dim3 gcol((L.Kpad + 255) / 256, NN);
    im2col_kernel<<<gcol, 256, 0, stream>>>(L.in, COL, L.Cin, L.Kpad);
    dim3 gw((L.Kpad + 255) / 256, L.CoutPad);
    wcvt_kernel<<<gw, 256, 0, stream>>>(L.w, WB, Ktot, L.Kpad, L.Cout);
    dim3 gg(NN / 128, L.CoutPad / 32);  // 8 waves/block, 16x32 tile per wave
    gemm_wmma_kernel<<<gg, 256, 0, stream>>>(COL, WB, L.out, L.st, L.Kpad, L.Cout);
    int total = L.Cout * NN;
    ln_relu_kernel<<<(total + 255) / 256, 256, 0, stream>>>(L.out, L.lw, L.lb, L.st, total);
  }

  // CAM (tiny)
  gram_kernel<<<289, 256, 0, stream>>>(H4, GRAM);
  cam_softmax_kernel<<<1, 32, 0, stream>>>(GRAM, ATTN);

  // PAM: QKV projection + fused flash attention (WMMA)
  qkv_kernel<<<NN / 256, 256, 0, stream>>>(H4, qw, qb, kw, kb, vw, vb, QT, KT, VB);
  flash_pam_kernel<<<NN / 128, 256, 0, stream>>>(QT, KT, VB, PAM);

  // y = cam + pam + h, conv5, LN+ReLU, conv6
  combine_conv5_kernel<<<NN / 256, 256, 0, stream>>>(H4, ATTN, PAM, camg, pamg, w5, C5, STATS + 8);
  ln_relu_kernel<<<(17 * NN + 255) / 256, 256, 0, stream>>>(C5, l5w, l5b, STATS + 8, 17 * NN);
  conv6_kernel<<<NN / 256, 256, 0, stream>>>(C5, w6, out);
}
